// SGA_Module_63093069578402
// MI455X (gfx1250) — matrix operand order repaired and checked by reading, not compiled, not run on hardware
//
#include <hip/hip_runtime.h>

// ---------------------------------------------------------------------------
// SGA attention module for MI455X (gfx1250, wave32, WMMA)
// B=4, C=64, N=4096, T=16, D=C*T=1024
// ws layout (f16): Qt (B,N,D) | Kt (B,N,D) | V (B,D,N) | S/P (B,N,N)
// ---------------------------------------------------------------------------

typedef __attribute__((ext_vector_type(16))) _Float16 v16h;
typedef __attribute__((ext_vector_type(8)))  _Float16 v8h;
typedef __attribute__((ext_vector_type(8)))  float    v8f;
typedef __attribute__((ext_vector_type(4)))  int      v4i;

#define NB   4
#define CC   64
#define NN   4096
#define TT   16
#define DD   1024
#define LDT  40   // padded LDS row stride (halfwords) -> conflict-free b128 reads

#if __has_builtin(__builtin_amdgcn_global_load_async_to_lds_b128)
#define HAVE_ASYNC_LDS 1
#else
#define HAVE_ASYNC_LDS 0
#endif

static __device__ inline void wait_asynccnt0() {
#if __has_builtin(__builtin_amdgcn_s_wait_asynccnt)
    __builtin_amdgcn_s_wait_asynccnt(0);
#else
    asm volatile("s_wait_asynccnt 0x0" ::: "memory");
#endif
}

// one 16B global -> LDS async DMA (ASYNCcnt tracked)
static __device__ inline void async_copy16(const _Float16* g, _Float16* l) {
#if HAVE_ASYNC_LDS
    __builtin_amdgcn_global_load_async_to_lds_b128((v4i*)g, (v4i*)l, 0, 0);
#else
    *(float4*)l = *(const float4*)g;
#endif
}

// ---- fragment builders (frag_a: CDNA5 ISA 16-bit A/B VGPR layout; frag_b: contiguous k) ----------
// A 16x32 f16: lane half h: v0-3 = K(8h..8h+7), v4-7 = K(16+8h..16+8h+7)
static __device__ inline v16h frag_a(const _Float16* p) {
    v8h lo = *(const v8h*)(p);        // K = 8h .. 8h+7
    v8h hi = *(const v8h*)(p + 16);   // K = 16+8h .. 16+8h+7
    return __builtin_shufflevector(lo, hi, 0,1,2,3,4,5,6,7,8,9,10,11,12,13,14,15);
}
// B 32x16 f16 in this helper's contiguous k order (ISA order is frag_a's): lane half h: v0-7 = K(16h .. 16h+15) (contiguous)
static __device__ inline v16h frag_b(const _Float16* p) {
    v8h lo = *(const v8h*)(p);        // K = 16h .. 16h+7
    v8h hi = *(const v8h*)(p + 8);    // K = 16h+8 .. 16h+15
    return __builtin_shufflevector(lo, hi, 0,1,2,3,4,5,6,7,8,9,10,11,12,13,14,15);
}

// ---------------------------------------------------------------------------
// Kernel 1: QKV 1x1-conv projection.
//   q[b,o,n,t] = sum_c wq[o,c] * x[b,c,n,t] + bq[o]   (same for k,v)
//   flat f = o*65536 + jhi*4096 + nn  ->  dd = o*16+jhi, col = nn
//   Writes Qt/Kt at [b][nn][dd] (N,D) and V at [b][dd][nn] (D,N), all f16.
// ---------------------------------------------------------------------------
__global__ __launch_bounds__(256) void sga_qkv_proj(
    const float* __restrict__ x,
    const float* __restrict__ wq, const float* __restrict__ bq,
    const float* __restrict__ wk, const float* __restrict__ bk,
    const float* __restrict__ wv, const float* __restrict__ bv,
    _Float16* __restrict__ Qt, _Float16* __restrict__ Kt,
    _Float16* __restrict__ V)
{
    __shared__ _Float16 wqs[64 * 64], wks[64 * 64], wvs[64 * 64];
    __shared__ float    bqs[64], bks[64], bvs[64];
    __shared__ _Float16 xs[64 * 256];   // [c][jhi*16 + nn]

    const int t   = threadIdx.x;
    const int b   = blockIdx.y;
    const int nn0 = blockIdx.x * 16;

#pragma unroll
    for (int i = 0; i < 16; ++i) {
        int e = t + 256 * i;
        wqs[e] = (_Float16)wq[e];
        wks[e] = (_Float16)wk[e];
        wvs[e] = (_Float16)wv[e];
    }
    if (t < 64) { bqs[t] = bq[t]; bks[t] = bk[t]; bvs[t] = bv[t]; }

    // stage x tile: c in [0,64), j = jhi*4096 + nn0 + nn  (coalesced over nn)
    for (int i = 0; i < 64; ++i) {
        int e   = i * 256 + t;
        int c   = e >> 8;
        int jq  = e & 255;            // jhi*16 + nn
        int jhi = jq >> 4;
        int nnl = jq & 15;
        size_t gi = ((size_t)(b * 64 + c) << 16) + (size_t)jhi * 4096 + nn0 + nnl;
        xs[e] = (_Float16)x[gi];
    }
    __syncthreads();

    const int jhi  = t & 15;          // consecutive threads -> consecutive dd
    const int nnl  = t >> 4;
    const int xoff = jhi * 16 + nnl;
    const size_t qkbase = ((size_t)b * NN + nn0 + nnl) * DD;
    const size_t vbase  = (size_t)b * DD * NN + (size_t)(nn0 + nnl);

    for (int o = 0; o < 64; ++o) {
        const _Float16* wqr = &wqs[o * 64];
        const _Float16* wkr = &wks[o * 64];
        const _Float16* wvr = &wvs[o * 64];
        float aq = 0.f, ak = 0.f, av = 0.f;
#pragma unroll 16
        for (int c = 0; c < 64; ++c) {
            float xv = (float)xs[c * 256 + xoff];
            aq += (float)wqr[c] * xv;
            ak += (float)wkr[c] * xv;
            av += (float)wvr[c] * xv;
        }
        aq += bqs[o]; ak += bks[o]; av += bvs[o];
        const int dd = o * 16 + jhi;
        Qt[qkbase + dd] = (_Float16)aq;
        Kt[qkbase + dd] = (_Float16)ak;
        V[vbase + (size_t)dd * NN] = (_Float16)av;
    }
}

// ---------------------------------------------------------------------------
// Kernels 2 & 4: NT-GEMM  C[m][n] = sum_k A[m][k] * B[n][k]
//   128x128 block tile, 8 waves (2x4), each wave 64x32 via 4x2 WMMA tiles,
//   double-buffered LDS filled by GLOBAL_LOAD_ASYNC_TO_LDS_B128 (ASYNCcnt),
//   v_wmma_f32_16x16x32_f16 mainloop, all fragments preloaded per k-step.
// EPI==0: C = f16 store (energy).  EPI==1: C = gamma*acc + X (float out).
// ---------------------------------------------------------------------------
template <int EPI, int KTOT>
__global__ __launch_bounds__(256) void sga_gemm_nt(
    const _Float16* __restrict__ Ag, const _Float16* __restrict__ Bg,
    int ldA, int ldB, long strideA, long strideB,
    void* __restrict__ Cg, int ldC, long strideC,
    const float* __restrict__ Xres, long strideX,
    const float* __restrict__ gamma)
{
    __shared__ _Float16 As[2][128 * LDT];
    __shared__ _Float16 Bs[2][128 * LDT];

    const int t  = threadIdx.x;
    const int z  = blockIdx.z;
    const int m0 = blockIdx.x * 128;
    const int n0 = blockIdx.y * 128;

    const _Float16* A = Ag + (size_t)z * strideA;
    const _Float16* B = Bg + (size_t)z * strideB;

    const int lane = t & 31;
    const int wid  = t >> 5;
    const int half = lane >> 4;
    const int lr   = lane & 15;
    const int wm   = (wid & 1) * 64;   // wave M offset (2 waves over M)
    const int wn   = (wid >> 1) * 32;  // wave N offset (4 waves over N)

    v8f acc[4][2];
#pragma unroll
    for (int i = 0; i < 4; ++i)
#pragma unroll
        for (int j = 0; j < 2; ++j) acc[i][j] = (v8f){0.f,0.f,0.f,0.f,0.f,0.f,0.f,0.f};

    // async DMA one 128x32 tile of A and B into LDS buffer `buf`
    auto load_tiles = [&](int buf, int kk) {
#pragma unroll
        for (int r = 0; r < 2; ++r) {
            int idx = t + 256 * r;
            int row = idx >> 2;
            int ch  = (idx & 3) * 8;   // 16B chunk within 32-halfword row
            async_copy16(A + (size_t)(m0 + row) * ldA + kk + ch,
                         &As[buf][row * LDT + ch]);
            async_copy16(B + (size_t)(n0 + row) * ldB + kk + ch,
                         &Bs[buf][row * LDT + ch]);
        }
    };

    load_tiles(0, 0);
    wait_asynccnt0();
    __syncthreads();

#pragma unroll 2
    for (int kk = 0; kk < KTOT; kk += 32) {
        const int buf = (kk >> 5) & 1;
        if (kk + 32 < KTOT) load_tiles(buf ^ 1, kk + 32);   // overlaps with WMMAs

        // preload all fragments so the ds_loads pipeline and WMMAs go back-to-back
        v16h bfr[2], afr[4];
#pragma unroll
        for (int sn = 0; sn < 2; ++sn)
            bfr[sn] = frag_a(&Bs[buf][(wn + sn * 16 + lr) * LDT + half * 8]);
#pragma unroll
        for (int sm = 0; sm < 4; ++sm)
            afr[sm] = frag_a(&As[buf][(wm + sm * 16 + lr) * LDT + half * 8]);

#pragma unroll
        for (int sm = 0; sm < 4; ++sm)
#pragma unroll
            for (int sn = 0; sn < 2; ++sn)
                acc[sm][sn] = __builtin_amdgcn_wmma_f32_16x16x32_f16(
                    false, afr[sm], false, bfr[sn], (short)0, acc[sm][sn], false, false);

        wait_asynccnt0();   // own async fills of buf^1 done
        __syncthreads();    // all waves done reading buf / filling buf^1
    }

    // epilogue: lane holds C[m = i + 8*half][n = lr] per 16x16 tile
    if constexpr (EPI == 0) {
        _Float16* C = (_Float16*)Cg + (size_t)z * strideC;
#pragma unroll
        for (int sm = 0; sm < 4; ++sm)
#pragma unroll
            for (int sn = 0; sn < 2; ++sn)
#pragma unroll
                for (int i = 0; i < 8; ++i) {
                    int row = m0 + wm + sm * 16 + half * 8 + i;
                    int col = n0 + wn + sn * 16 + lr;
                    C[(size_t)row * ldC + col] = (_Float16)acc[sm][sn][i];
                }
    } else {
        const float g = gamma[0];
        float* C = (float*)Cg + (size_t)z * strideC;
        const float* X = Xres + (size_t)z * strideX;
#pragma unroll
        for (int sm = 0; sm < 4; ++sm)
#pragma unroll
            for (int sn = 0; sn < 2; ++sn)
#pragma unroll
                for (int i = 0; i < 8; ++i) {
                    int row = m0 + wm + sm * 16 + half * 8 + i;
                    int col = n0 + wn + sn * 16 + lr;
                    size_t o = (size_t)row * ldC + col;
                    C[o] = g * acc[sm][sn][i] + X[o];
                }
    }
}

// ---------------------------------------------------------------------------
// Kernel 3: in-place row softmax over 4096 f16 values (one block per row).
// ---------------------------------------------------------------------------
__global__ __launch_bounds__(256) void sga_softmax(_Float16* __restrict__ S)
{
    const int t = threadIdx.x;
    _Float16* row = S + (size_t)blockIdx.x * NN;
    __shared__ float red[8];

    v8h a  = *(const v8h*)(row + t * 16);
    v8h b2 = *(const v8h*)(row + t * 16 + 8);
    float vals[16];
#pragma unroll
    for (int i = 0; i < 8; ++i) { vals[i] = (float)a[i]; vals[8 + i] = (float)b2[i]; }

    float m = -3.0e30f;
#pragma unroll
    for (int i = 0; i < 16; ++i) m = fmaxf(m, vals[i]);
#pragma unroll
    for (int off = 16; off > 0; off >>= 1) m = fmaxf(m, __shfl_xor(m, off));
    if ((t & 31) == 0) red[t >> 5] = m;
    __syncthreads();
    float rm = red[0];
#pragma unroll
    for (int i = 1; i < 8; ++i) rm = fmaxf(rm, red[i]);

    float s = 0.f;
#pragma unroll
    for (int i = 0; i < 16; ++i) { vals[i] = __expf(vals[i] - rm); s += vals[i]; }
#pragma unroll
    for (int off = 16; off > 0; off >>= 1) s += __shfl_xor(s, off);
    __syncthreads();
    if ((t & 31) == 0) red[t >> 5] = s;
    __syncthreads();
    float rs = 0.f;
#pragma unroll
    for (int i = 0; i < 8; ++i) rs += red[i];
    const float inv = 1.0f / rs;

#pragma unroll
    for (int i = 0; i < 8; ++i) {
        a[i]  = (_Float16)(vals[i] * inv);
        b2[i] = (_Float16)(vals[8 + i] * inv);
    }
    *(v8h*)(row + t * 16)     = a;
    *(v8h*)(row + t * 16 + 8) = b2;
}

// ---------------------------------------------------------------------------
extern "C" void kernel_launch(void* const* d_in, const int* in_sizes, int n_in,
                              void* d_out, int out_size, void* d_ws, size_t ws_size,
                              hipStream_t stream)
{
    const float* x     = (const float*)d_in[0];
    const float* wq    = (const float*)d_in[1];
    const float* bq    = (const float*)d_in[2];
    const float* wk    = (const float*)d_in[3];
    const float* bk    = (const float*)d_in[4];
    const float* wv    = (const float*)d_in[5];
    const float* bv    = (const float*)d_in[6];
    const float* gamma = (const float*)d_in[7];
    float* out = (float*)d_out;

    const size_t perQK = (size_t)NN * DD;       // 4,194,304 elems per batch
    const size_t perS  = (size_t)NN * NN;       // 16,777,216 elems per batch

    _Float16* Qt = (_Float16*)d_ws;             // (B,N,D) f16
    _Float16* Kt = Qt + (size_t)NB * perQK;     // (B,N,D) f16
    _Float16* V  = Kt + (size_t)NB * perQK;     // (B,D,N) f16
    _Float16* S  = V  + (size_t)NB * perQK;     // (B,N,N) f16 (energy -> probs)

    // 1) QKV projection
    sga_qkv_proj<<<dim3(NN / 16, NB), 256, 0, stream>>>(
        x, wq, bq, wk, bk, wv, bv, Qt, Kt, V);

    // 2) energy = Qt * Kt^T   (M=N=4096, K=1024)
    sga_gemm_nt<0, DD><<<dim3(NN / 128, NN / 128, NB), 256, 0, stream>>>(
        Qt, Kt, DD, DD, (long)perQK, (long)perQK,
        (void*)S, NN, (long)perS, nullptr, 0, nullptr);

    // 3) softmax rows in place
    sga_softmax<<<NB * NN, 256, 0, stream>>>(S);

    // 4) out = gamma * (V * P^T) + x   (M=1024, N=4096, K=4096)
    sga_gemm_nt<1, NN><<<dim3(DD / 128, NN / 128, NB), 256, 0, stream>>>(
        V, S, NN, NN, (long)perQK, (long)perS,
        (void*)out, NN, (long)perQK, x, (long)perQK, gamma);
}
